// BasicBlock_7911329759668
// MI455X (gfx1250) — matrix operand order repaired and checked by reading, not compiled, not run on hardware
//
#include <hip/hip_runtime.h>
#include <hip/hip_bf16.h>

// ---------------------------------------------------------------------------
// Sparse BasicBlock (B=32, C=256, H=W=56) as two implicit-GEMM 3x3 convs via
// v_wmma_f32_16x16x32_bf16. Activations staged block-wide into LDS with
// GLOBAL_LOAD_ASYNC_TO_LDS_B128 (ASYNCcnt, branch-free clamped addressing),
// WMMA fed from ds_load. All mask/BN/ReLU/vector glue fused into epilogues.
// ---------------------------------------------------------------------------

typedef __attribute__((ext_vector_type(16))) __bf16 v16bf;
typedef __attribute__((ext_vector_type(8)))  __bf16 v8bf;
typedef __attribute__((ext_vector_type(8)))  float  v8f;
typedef int vint4 __attribute__((vector_size(16)));

#define CCH   256
#define HDIM  56
#define WDIM  56
#define PIX   3136        // 56*56
#define NPIX  100352      // 32*3136

// ---------------- workspace layout (bytes) ----------------
#define XM_OFF   ((size_t)0)                          // bf16 NHWC x*mask_dilate
#define MID_OFF  ((size_t)51380224)                   // bf16 NHWC mid activation
#define W1F_OFF  ((size_t)102760448)                  // packed A-fragments conv1
#define W2F_OFF  ((size_t)103940096)                  // packed A-fragments conv2
#define BN_OFF   ((size_t)105119744)                  // inv1|beta1|inv2|beta2

#if __has_builtin(__builtin_amdgcn_global_load_async_to_lds_b128)
#define HAVE_ASYNC_LDS 1
#else
#define HAVE_ASYNC_LDS 0
#endif

__device__ __forceinline__ void async_g2l_b128(const char* g, char* l)
{
#if HAVE_ASYNC_LDS
    __builtin_amdgcn_global_load_async_to_lds_b128(
        (__attribute__((address_space(1))) vint4*)(void*)(const_cast<char*>(g)),
        (__attribute__((address_space(3))) vint4*)l, 0, 0);
#else
    *(vint4*)l = *(const vint4*)g;   // fallback: vmem load + ds_store
#endif
}

__device__ __forceinline__ void wait_async_lds()
{
#if HAVE_ASYNC_LDS
#if __has_builtin(__builtin_amdgcn_s_wait_asynccnt)
    __builtin_amdgcn_s_wait_asynccnt(0);
#else
    asm volatile("s_wait_asynccnt 0" ::: "memory");
#endif
#endif
}

// ---------------------------------------------------------------------------
__global__ __launch_bounds__(256) void bnprep_kernel(
    const float* g1, const float* b1, const float* m1, const float* v1,
    const float* g2, const float* b2, const float* m2, const float* v2,
    float* out)
{
    int c = threadIdx.x;
    float inv1 = rsqrtf(v1[c] + 1e-5f) * g1[c];
    out[c]       = inv1;
    out[256 + c] = b1[c] - m1[c] * inv1;
    float inv2 = rsqrtf(v2[c] + 1e-5f) * g2[c];
    out[512 + c] = inv2;
    out[768 + c] = b2[c] - m2[c] * inv2;
}

// ---------------------------------------------------------------------------
// Pack [O,I,3,3] f32 weights into per-lane WMMA A-fragments (contiguous K
// layout: lane<16 M=lane K={0..15}; lane>=16 M=lane-16
// K={16..31}). Fragment id f = ((kh*3+kw)*8 + kc)*16 + mt, 1024 B each.
// ---------------------------------------------------------------------------
__global__ __launch_bounds__(256) void pack_w_kernel(const float* w, __bf16* frags)
{
    int wid  = threadIdx.x >> 5;
    int lane = threadIdx.x & 31;
    int f    = blockIdx.x * 8 + wid;
    int mt   = f & 15;
    int kc   = (f >> 4) & 7;
    int kk   = f >> 7;
    int kh   = kk / 3, kw = kk % 3;
    int hi   = (lane >> 4) & 1;
    int o    = mt * 16 + (lane & 15);
    __bf16* dst = frags + (size_t)f * 512 + (size_t)lane * 16;
#pragma unroll
    for (int e = 0; e < 16; ++e) {
        int kl = e + hi * 16;
        int i  = kc * 32 + kl;
        dst[e] = (__bf16)w[(((size_t)o * CCH + i) * 3 + kh) * 3 + kw];
    }
}

// ---------------------------------------------------------------------------
// NCHW f32 -> NHWC bf16 transpose of (x * mask_dilate), LDS-tiled 32x32.
// ---------------------------------------------------------------------------
__global__ __launch_bounds__(256) void stage0_kernel(
    const float* __restrict__ x, const float* __restrict__ md, __bf16* __restrict__ xm)
{
    __shared__ float tile[32][33];
    int pblk = blockIdx.x * 32;
    int cblk = blockIdx.y * 32;
    int tx = threadIdx.x, ty = threadIdx.y;
    int b  = pblk / PIX;
    int p0 = pblk - b * PIX;
    const float* xbase = x + (size_t)b * (CCH * PIX) + (size_t)cblk * PIX + p0;
#pragma unroll
    for (int i = 0; i < 4; ++i) {
        int c = ty + i * 8;
        tile[c][tx] = xbase[(size_t)c * PIX + tx];
    }
    __syncthreads();
#pragma unroll
    for (int i = 0; i < 4; ++i) {
        int pp = ty + i * 8;
        float m = md[pblk + pp];
        xm[(size_t)(pblk + pp) * CCH + cblk + tx] = (__bf16)(tile[tx][pp] * m);
    }
}

// ---------------------------------------------------------------------------
// Implicit-GEMM 3x3 conv. Block = 8 waves = 64 out-ch x 128 pixels.
// LDS holds 256 flat-pixel rows [P0-64, P0+191] of NHWC bf16 (128 KB),
// filled with branch-free async global->LDS b128 copies (row index clamped:
// out-of-range rows carry junk but are never consumed — tap validity masks
// them). 9 taps x 8 kc chunks of WMMA fed entirely from ds_load.
// Wave: 16 ch x 64 px (4 accumulators).
// ---------------------------------------------------------------------------
template <int STAGE>
__global__ __launch_bounds__(256) void conv_wmma_kernel(
    const __bf16* __restrict__ in_nhwc,
    const __bf16* __restrict__ wfrag,
    const float* __restrict__ inv, const float* __restrict__ beta,
    const float* __restrict__ mask, const float* __restrict__ maskd,
    const float* __restrict__ vec,
    const float* __restrict__ x_nchw,
    float* __restrict__ out_f32,
    __bf16* __restrict__ mid)
{
    extern __shared__ char smem[];            // 131072 bytes = 256 rows * 512 B

    const int tid   = threadIdx.x;
    const int lane  = tid & 31;
    const int wid   = tid >> 5;
    const int hi    = lane >> 4;
    const int nlane = lane & 15;
    const int m_sub = wid & 3;
    const int n_sub = wid >> 2;
    const int m_base    = blockIdx.y * 64 + m_sub * 16;
    const int mt_global = blockIdx.y * 4 + m_sub;
    const int P0        = blockIdx.x * 128;
    const int wave_pb   = P0 + n_sub * 64;

    // ---- stage activation rows [P0-64, P0+191] into LDS, branch-free ----
    {
        const char* inb = (const char*)in_nhwc;
        const int base_row = P0 - 64;
#pragma unroll
        for (int i = 0; i < 32; ++i) {
            int chunk = tid + i * 256;                 // 0..8191 (16 B each)
            int grow  = base_row + (chunk >> 5);       // 32 chunks per 512 B row
            grow = min(max(grow, 0), NPIX - 1);        // clamp instead of branch
            async_g2l_b128(inb + (size_t)grow * 512 + (size_t)(chunk & 31) * 16,
                           smem + (size_t)chunk * 16);
        }
    }

    v8f acc[4] = {};

    int pfl[4], pb[4], ph[4], pw[4], rel[4];
#pragma unroll
    for (int t = 0; t < 4; ++t) {
        int p  = wave_pb + t * 16 + nlane;
        pfl[t] = p;
        int b  = p / PIX;
        int r  = p - b * PIX;
        pb[t]  = b;
        ph[t]  = r / WDIM;
        pw[t]  = r - (r / WDIM) * WDIM;
        rel[t] = p - P0 + 64;                          // LDS row of this pixel
    }

    wait_async_lds();
    __syncthreads();

    const char* wfb = (const char*)wfrag;

#pragma unroll
    for (int tap = 0; tap < 9; ++tap) {
        const int kh = tap / 3, kw = tap % 3;
        const int dflat = (kh - 1) * WDIM + (kw - 1);  // flat-pixel shift of this tap
        const char* bbase[4];
        bool val[4];
#pragma unroll
        for (int t = 0; t < 4; ++t) {
            int hh = ph[t] + kh - 1;
            int ww = pw[t] + kw - 1;
            val[t]   = ((unsigned)hh < HDIM) && ((unsigned)ww < WDIM);
            bbase[t] = smem + (size_t)(rel[t] + dflat) * 512 + hi * 32;
        }
        const char* afrag = wfb +
            ((size_t)(tap * 8) * 16 + mt_global) * 1024 + (size_t)lane * 32;
#pragma unroll
        for (int kc = 0; kc < 8; ++kc) {
            v16bf a = *(const v16bf*)(afrag + (size_t)kc * 16384);
#pragma unroll
            for (int t = 0; t < 4; ++t) {
                v16bf bv = {};
                if (val[t]) bv = *(const v16bf*)(bbase[t] + kc * 64);  // ds_load
                acc[t] = __builtin_amdgcn_wmma_f32_16x16x32_bf16(
                    false, a, false, bv, (short)0, acc[t], false, false);
            }
        }
    }

    // ---------------- fused epilogue ----------------
#pragma unroll
    for (int t = 0; t < 4; ++t) {
        int   pflat = pfl[t];
        float mv    = mask[pflat];
        if (STAGE == 1) {
            float mdv = maskd[pflat];
            v8bf  pk;
#pragma unroll
            for (int r = 0; r < 8; ++r) {
                int   ch = m_base + hi * 8 + r;         // 8 consecutive channels
                float h  = acc[t][r] * mv;              // conv1(..)*mask
                h = h * inv[ch] + beta[ch];             // bn1
                h = fmaxf(h, 0.0f) * mdv;               // relu * mask_dilate
                h = h * mv * vec[pb[t] * CCH + ch];     // * mask * vector
                pk[r] = (__bf16)h;
            }
            // one contiguous 16 B store (channels m_base+hi*8 .. +7)
            *(v8bf*)(mid + (size_t)pflat * CCH + m_base + hi * 8) = pk;
        } else {
            int pimg = pfl[t] - pb[t] * PIX;
#pragma unroll
            for (int r = 0; r < 8; ++r) {
                int   ch = m_base + hi * 8 + r;
                float h  = acc[t][r] * mv;              // conv2(..)*mask
                h = h * inv[ch] + beta[ch];             // bn2
                h = h * mv;                             // * mask
                size_t oidx = ((size_t)pb[t] * CCH + ch) * PIX + pimg;
                out_f32[oidx] = fmaxf(x_nchw[oidx] + h * mv, 0.0f);
            }
        }
    }
}

// ---------------------------------------------------------------------------
extern "C" void kernel_launch(void* const* d_in, const int* in_sizes, int n_in,
                              void* d_out, int out_size, void* d_ws, size_t ws_size,
                              hipStream_t stream)
{
    (void)in_sizes; (void)n_in; (void)out_size; (void)ws_size;
    const float* x      = (const float*)d_in[0];
    const float* mask   = (const float*)d_in[1];
    const float* maskd  = (const float*)d_in[2];
    const float* vec    = (const float*)d_in[3];
    const float* w1     = (const float*)d_in[4];
    const float* bn1_g  = (const float*)d_in[5];
    const float* bn1_b  = (const float*)d_in[6];
    const float* bn1_m  = (const float*)d_in[7];
    const float* bn1_v  = (const float*)d_in[8];
    const float* w2     = (const float*)d_in[9];
    const float* bn2_g  = (const float*)d_in[10];
    const float* bn2_b  = (const float*)d_in[11];
    const float* bn2_m  = (const float*)d_in[12];
    const float* bn2_v  = (const float*)d_in[13];
    float* out = (float*)d_out;

    char* ws = (char*)d_ws;
    __bf16* xm   = (__bf16*)(ws + XM_OFF);
    __bf16* mid  = (__bf16*)(ws + MID_OFF);
    __bf16* w1f  = (__bf16*)(ws + W1F_OFF);
    __bf16* w2f  = (__bf16*)(ws + W2F_OFF);
    float*  bnc  = (float*)(ws + BN_OFF);
    const float* inv1  = bnc;
    const float* beta1 = bnc + 256;
    const float* inv2  = bnc + 512;
    const float* beta2 = bnc + 768;

    bnprep_kernel<<<1, 256, 0, stream>>>(bn1_g, bn1_b, bn1_m, bn1_v,
                                         bn2_g, bn2_b, bn2_m, bn2_v, bnc);
    pack_w_kernel<<<144, 256, 0, stream>>>(w1, w1f);
    pack_w_kernel<<<144, 256, 0, stream>>>(w2, w2f);
    stage0_kernel<<<dim3(NPIX / 32, CCH / 32), dim3(32, 8), 0, stream>>>(x, maskd, xm);

    const size_t lds_bytes = 131072;   // 256 rows * 512 B
    conv_wmma_kernel<1><<<dim3(NPIX / 128, CCH / 64), 256, lds_bytes, stream>>>(
        xm, w1f, inv1, beta1, mask, maskd, vec, nullptr, nullptr, mid);
    conv_wmma_kernel<2><<<dim3(NPIX / 128, CCH / 64), 256, lds_bytes, stream>>>(
        mid, w2f, inv2, beta2, mask, nullptr, nullptr, x, out, nullptr);
}
